// SSM_40578851012839
// MI455X (gfx1250) — matrix operand order repaired and checked by reading, not compiled, not run on hardware
//
#include <hip/hip_runtime.h>

typedef _Float16 h16;
typedef __attribute__((ext_vector_type(4)))  _Float16 v4h;
typedef __attribute__((ext_vector_type(8)))  _Float16 v8h;
typedef __attribute__((ext_vector_type(16))) _Float16 v16h;
typedef __attribute__((ext_vector_type(8)))  float    v8f;

namespace {
constexpr int kB   = 4;
constexpr int kL   = 8192;
constexpr int kD   = 1024;
constexpr int kN   = 1024;
constexpr int kM   = kB * kL;     // 32768 tokens
constexpr int kK   = kD;          // 1024 reduction dim
constexpr int kSteps = kK / 32;   // 32 K-chunks
constexpr int kCH  = 128;         // scan chunk length
constexpr int kNCH = kL / kCH;    // 64 chunks
constexpr float kEPS = 1e-5f;
}

__device__ __forceinline__ float sigmoidf_fast(float x) {
  return 1.0f / (1.0f + __expf(-x));
}

// ---- CDNA5 async copy: 16B global -> LDS per lane (ASYNCcnt tracked) ----
__device__ __forceinline__ void async_copy16(unsigned lds_off, const void* gptr) {
  asm volatile("global_load_async_to_lds_b128 %0, %1, off"
               :: "v"(lds_off), "v"((unsigned long long)(uintptr_t)gptr)
               : "memory");
}
__device__ __forceinline__ void wait_async_le1() {
  asm volatile("s_wait_asynccnt 0x1" ::: "memory");
}
__device__ __forceinline__ void wait_async_le0() {
  asm volatile("s_wait_asynccnt 0x0" ::: "memory");
}
// Generic pointer to LDS carries the LDS byte offset in its low 32 bits.
__device__ __forceinline__ unsigned lds_offset_of(const void* p) {
  return (unsigned)(uintptr_t)p;
}

// A-matrix fragment (16x32 f16) from an LDS tile of 32 rows x 32 halfs
// (row stride 32 halfs). Lane l<16: row rb+l, K {0..7,16..23};
// lane>=16: K {8..15,24..31}  (ISA 7.12.2 16-bit A layout).
__device__ __forceinline__ v16h load_a_frag_lds(const h16* S, int row_base, int lane) {
  const int r  = row_base + (lane & 15);
  const int kh = (lane >> 4) << 3;            // 0 or 8
  const h16* p = S + r * 32 + kh;
  v8h lo = *(const v8h*)(p);
  v8h hi = *(const v8h*)(p + 16);
  v16h f;
#pragma unroll
  for (int i = 0; i < 8; ++i) { f[i] = lo[i]; f[i + 8] = hi[i]; }
  return f;
}

// B-matrix fragment (32x16 f16) of W^T, W row-major (out x in).
// Lane l<16 holds col n0+l, K {0..7,16..23}; lane>=16: K {8..15,24..31} (rel. kc).
__device__ __forceinline__ v16h load_b_frag(const h16* __restrict__ W, int ld,
                                            int col_base, int kc, int lane) {
  const int n  = col_base + (lane & 15);
  const int ko = (lane >> 4) << 3;            // 0 or 8
  const h16* p = W + (size_t)n * ld + kc + ko;
  v8h lo = *(const v8h*)(p);
  v8h hi = *(const v8h*)(p + 16);
  v16h f;
#pragma unroll
  for (int i = 0; i < 8; ++i) { f[i] = lo[i]; f[i + 8] = hi[i]; }
  return f;
}

__device__ __forceinline__ v8f wmma16x16x32(v16h a, v16h b, v8f c) {
  return __builtin_amdgcn_wmma_f32_16x16x32_f16(
      /*neg_a=*/false, a, /*neg_b=*/false, b,
      /*c_mod=*/(short)0, c, /*reuse_a=*/false, /*reuse_b=*/false);
}

// ---------------------------------------------------------------- RMSNorm
__global__ void rmsnorm_to_f16(const float* __restrict__ x,
                               const float* __restrict__ nw,
                               h16* __restrict__ xn) {
  __shared__ float red[256];
  const int tid = threadIdx.x;
  const size_t rb = (size_t)blockIdx.x * kD;
  const float4 v = ((const float4*)(x + rb))[tid];
  red[tid] = v.x * v.x + v.y * v.y + v.z * v.z + v.w * v.w;
  __syncthreads();
#pragma unroll
  for (int s = 128; s > 0; s >>= 1) {
    if (tid < s) red[tid] += red[tid + s];
    __syncthreads();
  }
  const float inv = rsqrtf(red[0] * (1.0f / (float)kD) + kEPS);
  const float4 w = ((const float4*)nw)[tid];
  v4h o;
  o[0] = (h16)(v.x * w.x * inv);
  o[1] = (h16)(v.y * w.y * inv);
  o[2] = (h16)(v.z * w.z * inv);
  o[3] = (h16)(v.w * w.w * inv);
  ((v4h*)(xn + rb))[tid] = o;
}

// ------------------------------------------------------- weight f32->f16
__global__ void cvt_f32_to_f16(const float* __restrict__ src,
                               h16* __restrict__ dst, int n) {
  const int i = blockIdx.x * blockDim.x + threadIdx.x;
  if (i < n) dst[i] = (h16)src[i];
}

// ------------------------------------------- fused 4-way projection GEMM
// Block = 8 waves sharing one 32-row M-strip; A-tile (32x32 f16, 2KB) is
// staged to LDS with double-buffered GLOBAL_LOAD_ASYNC_TO_LDS_B128 by
// waves 0..3.  Each wave: 2 M-tiles x 1 N-tile x 4 weights (8 f32 accs).
// Epilogue fuses: a = sigm(gA+A_b), c = sigm(gC), u = sigm(gD)*gB.
__global__ void __launch_bounds__(256)
proj_gemm_wmma(const h16* __restrict__ xn,
               const h16* __restrict__ wA, const h16* __restrict__ wB,
               const h16* __restrict__ wC, const h16* __restrict__ wD,
               const float* __restrict__ A_b,
               h16* __restrict__ a16, h16* __restrict__ c16,
               float* __restrict__ u32) {
  __shared__ __align__(16) h16 smem[2][32 * 32];

  const int lane = threadIdx.x & 31;
  const int wave = threadIdx.x >> 5;
  const int mstrip = blockIdx.x >> 3;                 // 0..1023
  const int ntile  = ((blockIdx.x & 7) << 3) + wave;  // 0..63
  const int m0 = mstrip * 32;
  const int n0 = ntile * 16;

  // Async staging assignment: 128 lanes (waves 0..3) x 16B = 2KB chunk.
  const int q    = wave * 32 + lane;    // 0..255 (q<128 active)
  const int arow = q >> 2;              // 0..31
  const int aseg = q & 3;               // 16B segment within a 64B row
  const h16* agp = xn + (size_t)(m0 + arow) * kK + aseg * 8;
  const unsigned aoff = (unsigned)(arow * 64 + aseg * 16);  // bytes in tile
  const unsigned ldsb[2] = { lds_offset_of(&smem[0][0]),
                             lds_offset_of(&smem[1][0]) };

  if (wave < 4) async_copy16(ldsb[0] + aoff, agp);  // prologue: chunk 0

  const h16* Wp[4] = {wA, wB, wC, wD};
  v8f acc[2][4];
  v8f z = {};
#pragma unroll
  for (int i = 0; i < 2; ++i)
#pragma unroll
    for (int j = 0; j < 4; ++j) acc[i][j] = z;

  for (int i = 0; i < kSteps; ++i) {
    const int kc  = i * 32;
    const int cur = i & 1;
    const bool more = (i + 1 < kSteps);
    if (more && wave < 4)
      async_copy16(ldsb[1 - cur] + aoff, agp + kc + 32);
    if (more) wait_async_le1(); else wait_async_le0();
    __syncthreads();                       // chunk i visible in smem[cur]

    const v16h a0 = load_a_frag_lds(&smem[cur][0], 0,  lane);
    const v16h a1 = load_a_frag_lds(&smem[cur][0], 16, lane);
#pragma unroll
    for (int w = 0; w < 4; ++w) {
      const v16h b = load_b_frag(Wp[w], kK, n0, kc, lane);
      acc[0][w] = wmma16x16x32(a0, b, acc[0][w]);
      acc[1][w] = wmma16x16x32(a1, b, acc[1][w]);
    }
    __syncthreads();                       // done reading before re-issue
  }

  const int col = n0 + (lane & 15);
  const float bias = A_b[col];
#pragma unroll
  for (int mt = 0; mt < 2; ++mt) {
#pragma unroll
    for (int r = 0; r < 8; ++r) {
      const int row = m0 + mt * 16 + ((lane >> 4) << 3) + r;
      const size_t idx = (size_t)row * kN + col;
      const float av = sigmoidf_fast(acc[mt][0][r] + bias);
      const float bv = acc[mt][1][r];
      const float cv = sigmoidf_fast(acc[mt][2][r]);
      const float dv = sigmoidf_fast(acc[mt][3][r]);
      a16[idx] = (h16)av;
      c16[idx] = (h16)cv;
      u32[idx] = dv * bv;
    }
  }
}

// --------------------------------------------------- chunked scan pass 1
__global__ void scan_chunk_reduce(const h16* __restrict__ a16,
                                  const float* __restrict__ u32,
                                  float* __restrict__ carryA,
                                  float* __restrict__ carryH) {
  const int t = blockIdx.x;
  const int ngrp = t & 3;                 // kN/256 = 4 groups
  const int ch   = (t >> 2) & (kNCH - 1);
  const int b    = t >> 8;
  const int n = ngrp * 256 + threadIdx.x;
  size_t base = ((size_t)b * kL + (size_t)ch * kCH) * kN + n;
  float ap = 1.0f, hp = 0.0f;
#pragma unroll 4
  for (int l = 0; l < kCH; ++l) {
    const float a = (float)a16[base];
    const float u = u32[base];
    hp = a * hp + u;
    ap *= a;
    base += kN;
  }
  const size_t ci = ((size_t)b * kNCH + ch) * kN + n;
  carryA[ci] = ap;
  carryH[ci] = hp;
}

// --------------------------------------------------- chunked scan pass 2
__global__ void scan_carry(const float* __restrict__ carryA,
                           const float* __restrict__ carryH,
                           float* __restrict__ carryIn) {
  const int gid = blockIdx.x * blockDim.x + threadIdx.x;  // 0..B*N-1
  const int b = gid / kN;
  const int n = gid % kN;
  float h = 0.0f;
  for (int ch = 0; ch < kNCH; ++ch) {
    const size_t i = ((size_t)b * kNCH + ch) * kN + n;
    carryIn[i] = h;
    h = carryA[i] * h + carryH[i];
  }
}

// --------------------------------------------------- chunked scan pass 3
__global__ void scan_apply(const h16* __restrict__ a16,
                           const float* __restrict__ u32,
                           const h16* __restrict__ c16,
                           const float* __restrict__ carryIn,
                           h16* __restrict__ hg16) {
  const int t = blockIdx.x;
  const int ngrp = t & 3;
  const int ch   = (t >> 2) & (kNCH - 1);
  const int b    = t >> 8;
  const int n = ngrp * 256 + threadIdx.x;
  size_t base = ((size_t)b * kL + (size_t)ch * kCH) * kN + n;
  float h = carryIn[((size_t)b * kNCH + ch) * kN + n];
#pragma unroll 4
  for (int l = 0; l < kCH; ++l) {
    const float a = (float)a16[base];
    const float u = u32[base];
    h = a * h + u;
    hg16[base] = (h16)((float)c16[base] * h);
    base += kN;
  }
}

// ------------------------------------------ output GEMM + bias + residual
// Same LDS async staging of the shared A-strip; each wave: 2 M x 2 N tiles.
__global__ void __launch_bounds__(256)
out_gemm_wmma(const h16* __restrict__ hg16,
              const h16* __restrict__ wO,
              const float* __restrict__ out_b,
              const float* __restrict__ resid,
              float* __restrict__ y) {
  __shared__ __align__(16) h16 smem[2][32 * 32];

  const int lane = threadIdx.x & 31;
  const int wave = threadIdx.x >> 5;
  const int mstrip = blockIdx.x >> 2;                 // 0..1023
  const int nstrip = ((blockIdx.x & 3) << 3) + wave;  // 0..31
  const int m0 = mstrip * 32;
  const int n0 = nstrip * 32;

  const int q    = wave * 32 + lane;
  const int arow = q >> 2;
  const int aseg = q & 3;
  const h16* agp = hg16 + (size_t)(m0 + arow) * kN + aseg * 8;
  const unsigned aoff = (unsigned)(arow * 64 + aseg * 16);
  const unsigned ldsb[2] = { lds_offset_of(&smem[0][0]),
                             lds_offset_of(&smem[1][0]) };

  if (wave < 4) async_copy16(ldsb[0] + aoff, agp);

  v8f acc[2][2];
  v8f z = {};
#pragma unroll
  for (int i = 0; i < 2; ++i)
#pragma unroll
    for (int j = 0; j < 2; ++j) acc[i][j] = z;

  for (int i = 0; i < kSteps; ++i) {
    const int kc  = i * 32;
    const int cur = i & 1;
    const bool more = (i + 1 < kSteps);
    if (more && wave < 4)
      async_copy16(ldsb[1 - cur] + aoff, agp + kc + 32);
    if (more) wait_async_le1(); else wait_async_le0();
    __syncthreads();

    const v16h a0 = load_a_frag_lds(&smem[cur][0], 0,  lane);
    const v16h a1 = load_a_frag_lds(&smem[cur][0], 16, lane);
    const v16h b0 = load_b_frag(wO, kN, n0,      kc, lane);
    const v16h b1 = load_b_frag(wO, kN, n0 + 16, kc, lane);
    acc[0][0] = wmma16x16x32(a0, b0, acc[0][0]);
    acc[0][1] = wmma16x16x32(a0, b1, acc[0][1]);
    acc[1][0] = wmma16x16x32(a1, b0, acc[1][0]);
    acc[1][1] = wmma16x16x32(a1, b1, acc[1][1]);
    __syncthreads();
  }

#pragma unroll
  for (int nt = 0; nt < 2; ++nt) {
    const int col = n0 + nt * 16 + (lane & 15);
    const float ob = out_b[col];
#pragma unroll
    for (int mt = 0; mt < 2; ++mt) {
#pragma unroll
      for (int r = 0; r < 8; ++r) {
        const int row = m0 + mt * 16 + ((lane >> 4) << 3) + r;
        const size_t idx = (size_t)row * kD + col;
        y[idx] = acc[mt][nt][r] + ob + resid[idx];
      }
    }
  }
}

// ------------------------------------------------------------------------
extern "C" void kernel_launch(void* const* d_in, const int* in_sizes, int n_in,
                              void* d_out, int out_size, void* d_ws, size_t ws_size,
                              hipStream_t stream) {
  (void)in_sizes; (void)n_in; (void)out_size; (void)ws_size;
  const float* x      = (const float*)d_in[0];
  const float* norm_w = (const float*)d_in[1];
  const float* A_w    = (const float*)d_in[2];
  const float* A_b    = (const float*)d_in[3];
  const float* B_w    = (const float*)d_in[4];
  const float* C_w    = (const float*)d_in[5];
  const float* D_w    = (const float*)d_in[6];
  const float* out_w  = (const float*)d_in[7];
  const float* out_b  = (const float*)d_in[8];
  float* y = (float*)d_out;

  // Workspace carve-out (~397 MB total)
  char* p = (char*)d_ws;
  h16* xn16 = (h16*)p;        p += (size_t)kM * kD * 2;
  h16* wA16 = (h16*)p;        p += (size_t)kN * kK * 2;
  h16* wB16 = (h16*)p;        p += (size_t)kN * kK * 2;
  h16* wC16 = (h16*)p;        p += (size_t)kN * kK * 2;
  h16* wD16 = (h16*)p;        p += (size_t)kN * kK * 2;
  h16* wO16 = (h16*)p;        p += (size_t)kD * kN * 2;
  h16* a16  = (h16*)p;        p += (size_t)kM * kN * 2;
  h16* c16  = (h16*)p;        p += (size_t)kM * kN * 2;
  h16* hg16 = (h16*)p;        p += (size_t)kM * kN * 2;
  float* u32b    = (float*)p; p += (size_t)kM * kN * 4;
  float* carryA  = (float*)p; p += (size_t)kB * kNCH * kN * 4;
  float* carryH  = (float*)p; p += (size_t)kB * kNCH * kN * 4;
  float* carryIn = (float*)p; p += (size_t)kB * kNCH * kN * 4;

  // 1. RMSNorm + cast
  rmsnorm_to_f16<<<kM, 256, 0, stream>>>(x, norm_w, xn16);

  // 2. weight casts (recomputed each call; no cached state)
  const int wcnt = kN * kK;
  cvt_f32_to_f16<<<wcnt / 256, 256, 0, stream>>>(A_w,   wA16, wcnt);
  cvt_f32_to_f16<<<wcnt / 256, 256, 0, stream>>>(B_w,   wB16, wcnt);
  cvt_f32_to_f16<<<wcnt / 256, 256, 0, stream>>>(C_w,   wC16, wcnt);
  cvt_f32_to_f16<<<wcnt / 256, 256, 0, stream>>>(D_w,   wD16, wcnt);
  cvt_f32_to_f16<<<wcnt / 256, 256, 0, stream>>>(out_w, wO16, wcnt);

  // 3. fused 4-projection WMMA GEMM + activations (async LDS staging)
  proj_gemm_wmma<<<(kM / 32) * (kN / 16) / 8, 256, 0, stream>>>(
      xn16, wA16, wB16, wC16, wD16, A_b, a16, c16, u32b);

  // 4. chunked selective scan
  scan_chunk_reduce<<<kB * kNCH * (kN / 256), 256, 0, stream>>>(a16, u32b, carryA, carryH);
  scan_carry<<<(kB * kN) / 256, 256, 0, stream>>>(carryA, carryH, carryIn);
  scan_apply<<<kB * kNCH * (kN / 256), 256, 0, stream>>>(a16, u32b, c16, carryIn, hg16);

  // 5. output WMMA GEMM + bias + residual (async LDS staging)
  out_gemm_wmma<<<(kM / 32) * (kD / 32) / 8, 256, 0, stream>>>(
      hg16, wO16, out_b, x, y);
}
